// GAT_42786464202914
// MI455X (gfx1250) — matrix operand order repaired and checked by reading, not compiled, not run on hardware
//
#include <hip/hip_runtime.h>
#include <hip/hip_bf16.h>

#define HC 256          // H*C
#define HEADS 4
#define CH 64
#define FIN 128
#define NEG_SLOPE 0.2f
#define BN_EPS 1e-5f
#define NGRAPHS 64
#define MLP_HID 32

typedef _Float16 half_t;
typedef __attribute__((ext_vector_type(16))) _Float16 v16h;
typedef __attribute__((ext_vector_type(8)))  _Float16 v8h;
typedef __attribute__((ext_vector_type(8)))  float    v8f;

// ---------------- utility kernels ----------------
__global__ void k_fill_f32(float* __restrict__ p, float v, long n) {
  long i = (long)blockIdx.x * blockDim.x + threadIdx.x;
  if (i < n) p[i] = v;
}
__global__ void k_fill_u32(unsigned* __restrict__ p, unsigned v, long n) {
  long i = (long)blockIdx.x * blockDim.x + threadIdx.x;
  if (i < n) p[i] = v;
}
__global__ void k_f32_to_f16(const float* __restrict__ s, half_t* __restrict__ d, long n) {
  long i = (long)blockIdx.x * blockDim.x + threadIdx.x;
  if (i < n) d[i] = (half_t)s[i];
}
// W [K, 256] f32  ->  Wt [256, K] f16 (row n holds column n of W, contiguous in k)
__global__ void k_transpose_w(const float* __restrict__ W, half_t* __restrict__ Wt, int K) {
  int idx = blockIdx.x * blockDim.x + threadIdx.x;
  if (idx >= K * HC) return;
  int n = idx / K, k = idx - n * K;
  Wt[idx] = (half_t)W[(size_t)k * HC + n];
}

// ---------------- WMMA GEMM: C[N,256] = A[N,K] * W[K,256] ----------------
// A: row-major f16. Wt: [256,K] f16 (W transposed). 4 waves/block; wave w owns
// columns [64w, 64w+64); block owns 32 rows (two 16-row tiles per wave so each
// B operand feeds two WMMA chains). 16-bit A layout: lane(l) r=l&15, half=l>>4
// holds row r: elements 0..15 = K[k0+16*half .. +15].
// B layout: lane r holds column n=r, elements 0..15 = K[k0+16*half .. +15].
__device__ __forceinline__ v16h load_a16(const half_t* __restrict__ row, int k0, int half_) {
  v8h lo = *(const v8h*)(row + k0 + half_ * 16);
  v8h hi = *(const v8h*)(row + k0 + 8 + half_ * 16);
  v16h a;
#pragma unroll
  for (int i = 0; i < 8; ++i) { a[i] = lo[i]; a[i + 8] = hi[i]; }
  return a;
}
__device__ __forceinline__ v16h load_b16(const half_t* __restrict__ col, int k0, int half_) {
  return *(const v16h*)(col + k0 + half_ * 16);
}
#define WMMA_F16(A_, B_, C_) \
  __builtin_amdgcn_wmma_f32_16x16x32_f16(false, (A_), false, (B_), (short)0, (C_), false, false)

__global__ __launch_bounds__(128)
void k_gemm_wmma(const half_t* __restrict__ A, const half_t* __restrict__ Wt,
                 float* __restrict__ C, int N, int K) {
  const int lane  = threadIdx.x & 31;
  const int wave  = threadIdx.x >> 5;      // column group 0..3 (== head)
  const int half_ = lane >> 4;
  const int r     = lane & 15;
  const int row0  = blockIdx.x * 32;       // two 16-row tiles

  int ar0 = row0 + r;       if (ar0 >= N) ar0 = N - 1;  // clamped load rows
  int ar1 = row0 + 16 + r;  if (ar1 >= N) ar1 = N - 1;
  const half_t* Ap0 = A + (size_t)ar0 * K;
  const half_t* Ap1 = A + (size_t)ar1 * K;
  const int colbase = wave * 64;
  const half_t* B0 = Wt + (size_t)(colbase +  0 + r) * K;
  const half_t* B1 = Wt + (size_t)(colbase + 16 + r) * K;
  const half_t* B2 = Wt + (size_t)(colbase + 32 + r) * K;
  const half_t* B3 = Wt + (size_t)(colbase + 48 + r) * K;

  v8f c00 = {}, c01 = {}, c02 = {}, c03 = {};   // rows [row0,row0+16)
  v8f c10 = {}, c11 = {}, c12 = {}, c13 = {};   // rows [row0+16,row0+32)
  for (int k0 = 0; k0 < K; k0 += 32) {
    if (k0 + 32 < K) {
      __builtin_prefetch((const void*)(Ap0 + k0 + 32), 0, 3);
      __builtin_prefetch((const void*)(Ap1 + k0 + 32), 0, 3);
    }
    v16h a0 = load_a16(Ap0, k0, half_);
    v16h a1 = load_a16(Ap1, k0, half_);
    v16h b0 = load_b16(B0, k0, half_);
    v16h b1 = load_b16(B1, k0, half_);
    v16h b2 = load_b16(B2, k0, half_);
    v16h b3 = load_b16(B3, k0, half_);
    c00 = WMMA_F16(a0, b0, c00);  c10 = WMMA_F16(a1, b0, c10);
    c01 = WMMA_F16(a0, b1, c01);  c11 = WMMA_F16(a1, b1, c11);
    c02 = WMMA_F16(a0, b2, c02);  c12 = WMMA_F16(a1, b2, c12);
    c03 = WMMA_F16(a0, b3, c03);  c13 = WMMA_F16(a1, b3, c13);
  }
  // C/D layout: VGPR i -> row m = rowtile + 8*half + i, col n = colbase + 16*j + r
#pragma unroll
  for (int i = 0; i < 8; ++i) {
    int m0 = row0 + half_ * 8 + i;
    if (m0 < N) {
      float* crow = C + (size_t)m0 * HC + colbase + r;
      crow[0] = c00[i]; crow[16] = c01[i]; crow[32] = c02[i]; crow[48] = c03[i];
    }
    int m1 = row0 + 16 + half_ * 8 + i;
    if (m1 < N) {
      float* crow = C + (size_t)m1 * HC + colbase + r;
      crow[0] = c10[i]; crow[16] = c11[i]; crow[32] = c12[i]; crow[48] = c13[i];
    }
  }
}

// ---------------- per-node attention scores ----------------
__global__ void k_scores(const float* __restrict__ xp, const float* __restrict__ as,
                         const float* __restrict__ ad, float* __restrict__ es,
                         float* __restrict__ ed, int N) {
  int t = blockIdx.x * blockDim.x + threadIdx.x;
  if (t >= N * HEADS) return;
  int n = t >> 2, h = t & 3;
  const float4* xr  = (const float4*)(xp + (size_t)n * HC + h * CH);
  const float4* asr = (const float4*)(as + h * CH);
  const float4* adr = (const float4*)(ad + h * CH);
  float s = 0.f, d = 0.f;
#pragma unroll
  for (int i = 0; i < CH / 4; ++i) {
    float4 x = xr[i], A = asr[i], D = adr[i];
    s += x.x * A.x + x.y * A.y + x.z * A.z + x.w * A.w;
    d += x.x * D.x + x.y * D.y + x.z * D.z + x.w * D.w;
  }
  es[t] = s; ed[t] = d;
}

// order-preserving float<->uint map for atomic max
__device__ __forceinline__ unsigned ford(float f) {
  unsigned u = __float_as_uint(f);
  return (u & 0x80000000u) ? ~u : (u | 0x80000000u);
}
__device__ __forceinline__ float forddec(unsigned u) {
  return __uint_as_float((u & 0x80000000u) ? (u & 0x7fffffffu) : ~u);
}
__device__ __forceinline__ void edge_sd(const int* __restrict__ ei, int e, int E, int& s, int& d) {
  if (e < E) { s = ei[e]; d = ei[E + e]; } else { s = d = e - E; }   // self-loops appended
}

__global__ void k_edge_max(const int* __restrict__ ei, const float* __restrict__ es,
                           const float* __restrict__ ed, unsigned* __restrict__ mord,
                           int E, int N) {
  long t = (long)blockIdx.x * blockDim.x + threadIdx.x;
  if (t >= (long)(E + N) * HEADS) return;
  int e = (int)(t >> 2), h = (int)(t & 3), s, d;
  edge_sd(ei, e, E, s, d);
  float v = es[s * HEADS + h] + ed[d * HEADS + h];
  v = v > 0.f ? v : NEG_SLOPE * v;
  atomicMax(&mord[d * HEADS + h], ford(v));
}

__global__ void k_edge_exp(const int* __restrict__ ei, const float* __restrict__ es,
                           const float* __restrict__ ed, const unsigned* __restrict__ mord,
                           float* __restrict__ eexp, float* __restrict__ denom,
                           int E, int N) {
  long t = (long)blockIdx.x * blockDim.x + threadIdx.x;
  if (t >= (long)(E + N) * HEADS) return;
  int e = (int)(t >> 2), h = (int)(t & 3), s, d;
  edge_sd(ei, e, E, s, d);
  float v = es[s * HEADS + h] + ed[d * HEADS + h];
  v = v > 0.f ? v : NEG_SLOPE * v;
  float ee = __expf(v - forddec(mord[d * HEADS + h]));
  eexp[t] = ee;
  atomicAdd(&denom[d * HEADS + h], ee);
}

__global__ __launch_bounds__(HC)
void k_edge_scatter(const int* __restrict__ ei, const float* __restrict__ xp,
                    const float* __restrict__ eexp, const float* __restrict__ denom,
                    float* __restrict__ accum, int E, int N) {
  __shared__ int ss, sd;
  __shared__ float salpha[HEADS];
  int e = blockIdx.x, c = threadIdx.x;
  if (c == 0) { int s, d; edge_sd(ei, e, E, s, d); ss = s; sd = d; }
  __syncthreads();
  int d = sd;
  if (c < HEADS)
    salpha[c] = eexp[(size_t)e * HEADS + c] / (denom[d * HEADS + c] + 1e-16f);
  __syncthreads();
  float a = salpha[c >> 6];
  atomicAdd(&accum[(size_t)d * HC + c], xp[(size_t)ss * HC + c] * a);
}

// ---------------- bias + BN (+ReLU) ; writes f32 in place and f16 for next GEMM ----
__global__ void k_bn(float* __restrict__ accum, const float* __restrict__ b,
                     const float* __restrict__ g, const float* __restrict__ be,
                     const float* __restrict__ rm, const float* __restrict__ rv,
                     half_t* __restrict__ h16, long n, int relu) {
  long t = (long)blockIdx.x * blockDim.x + threadIdx.x;
  if (t >= n) return;
  int c = (int)(t & (HC - 1));
  float h = accum[t] + b[c];
  h = (h - rm[c]) * g[c] * rsqrtf(rv[c] + BN_EPS) + be[c];
  if (relu) h = fmaxf(h, 0.f);
  accum[t] = h;
  h16[t] = (half_t)h;
}

// ---------------- pooling + MLP head ----------------
__global__ __launch_bounds__(HC)
void k_pool(const float* __restrict__ h, const int* __restrict__ batch,
            float* __restrict__ pooled, float* __restrict__ cnt, int N) {
  int n = blockIdx.x, c = threadIdx.x;
  int b = batch[n];
  atomicAdd(&pooled[b * HC + c], h[(size_t)n * HC + c]);
  if (c == 0) atomicAdd(&cnt[b], 1.f);
}
__global__ void k_mlp1(const float* __restrict__ pooled, const float* __restrict__ cnt,
                       const float* __restrict__ Wm1, const float* __restrict__ bm1,
                       float* __restrict__ hid) {
  int t = blockIdx.x * blockDim.x + threadIdx.x;
  if (t >= NGRAPHS * MLP_HID) return;
  int gIdx = t >> 5, j = t & 31;
  float inv = 1.f / fmaxf(cnt[gIdx], 1.f);
  float s = 0.f;
  for (int k = 0; k < HC; ++k) s += pooled[gIdx * HC + k] * Wm1[k * MLP_HID + j];
  s = s * inv + bm1[j];
  hid[t] = fmaxf(s, 0.f);
}
__global__ void k_mlp2(const float* __restrict__ hid, const float* __restrict__ Wm2,
                       const float* __restrict__ bm2, float* __restrict__ out) {
  int gIdx = threadIdx.x;
  if (gIdx >= NGRAPHS) return;
  float s = 0.f;
  for (int j = 0; j < MLP_HID; ++j) s += hid[gIdx * MLP_HID + j] * Wm2[j];
  s += bm2[0];
  out[gIdx] = 1.f / (1.f + __expf(-s));
}

// ---------------- host launcher ----------------
extern "C" void kernel_launch(void* const* d_in, const int* in_sizes, int n_in,
                              void* d_out, int out_size, void* d_ws, size_t ws_size,
                              hipStream_t stream) {
  const float* x     = (const float*)d_in[0];
  const int*   ei    = (const int*)d_in[1];
  const int*   batch = (const int*)d_in[2];
  const int N = in_sizes[0] / FIN;
  const int E = in_sizes[1] / 2;
  const float *W[4], *as[4], *ad[4], *bb[4], *gg[4], *be[4], *rm[4], *rv[4];
  for (int l = 0; l < 4; ++l) {
    int base = 3 + 8 * l;
    W[l]  = (const float*)d_in[base + 0];
    as[l] = (const float*)d_in[base + 1];
    ad[l] = (const float*)d_in[base + 2];
    bb[l] = (const float*)d_in[base + 3];
    gg[l] = (const float*)d_in[base + 4];
    be[l] = (const float*)d_in[base + 5];
    rm[l] = (const float*)d_in[base + 6];
    rv[l] = (const float*)d_in[base + 7];
  }
  const float* Wm1 = (const float*)d_in[35];
  const float* bm1 = (const float*)d_in[36];
  const float* Wm2 = (const float*)d_in[37];
  const float* bm2 = (const float*)d_in[38];
  float* out = (float*)d_out;

  // workspace carving (256B aligned)
  size_t off = 0;
  auto carve = [&](size_t bytes) -> void* {
    void* p = (char*)d_ws + off;
    off = (off + bytes + 255) & ~(size_t)255;
    return p;
  };
  half_t*   h16    = (half_t*)  carve((size_t)N * HC * sizeof(half_t));
  half_t*   wt16   = (half_t*)  carve((size_t)HC * HC * sizeof(half_t));
  float*    xp     = (float*)   carve((size_t)N * HC * sizeof(float));
  float*    es     = (float*)   carve((size_t)N * HEADS * sizeof(float));
  float*    ed     = (float*)   carve((size_t)N * HEADS * sizeof(float));
  unsigned* mord   = (unsigned*)carve((size_t)N * HEADS * sizeof(unsigned));
  float*    denom  = (float*)   carve((size_t)N * HEADS * sizeof(float));
  float*    eexp   = (float*)   carve((size_t)(E + N) * HEADS * sizeof(float));
  float*    accum  = (float*)   carve((size_t)N * HC * sizeof(float));
  float*    pooled = (float*)   carve((size_t)NGRAPHS * HC * sizeof(float));
  float*    cnt    = (float*)   carve((size_t)NGRAPHS * sizeof(float));
  float*    hid    = (float*)   carve((size_t)NGRAPHS * MLP_HID * sizeof(float));

  const int T = 256;
  auto blocks = [&](long n) { return (unsigned)((n + T - 1) / T); };

  // initial input -> f16
  k_f32_to_f16<<<blocks((long)N * FIN), T, 0, stream>>>(x, h16, (long)N * FIN);

  const long nhc  = (long)N * HC;
  const long nh4  = (long)N * HEADS;
  const long et4  = (long)(E + N) * HEADS;

  for (int l = 0; l < 4; ++l) {
    const int K = (l == 0) ? FIN : HC;
    k_transpose_w<<<blocks((long)K * HC), T, 0, stream>>>(W[l], wt16, K);
    k_gemm_wmma<<<(unsigned)((N + 31) / 32), 128, 0, stream>>>(h16, wt16, xp, N, K);
    k_scores<<<blocks(nh4), T, 0, stream>>>(xp, as[l], ad[l], es, ed, N);
    k_fill_u32<<<blocks(nh4), T, 0, stream>>>(mord, 0u, nh4);
    k_fill_f32<<<blocks(nh4), T, 0, stream>>>(denom, 0.f, nh4);
    k_fill_f32<<<blocks(nhc), T, 0, stream>>>(accum, 0.f, nhc);
    k_edge_max<<<blocks(et4), T, 0, stream>>>(ei, es, ed, mord, E, N);
    k_edge_exp<<<blocks(et4), T, 0, stream>>>(ei, es, ed, mord, eexp, denom, E, N);
    k_edge_scatter<<<(unsigned)(E + N), HC, 0, stream>>>(ei, xp, eexp, denom, accum, E, N);
    k_bn<<<blocks(nhc), T, 0, stream>>>(accum, bb[l], gg[l], be[l], rm[l], rv[l],
                                        h16, nhc, (l < 3) ? 1 : 0);
  }

  // global mean pool + MLP head
  k_fill_f32<<<blocks(NGRAPHS * HC), T, 0, stream>>>(pooled, 0.f, NGRAPHS * HC);
  k_fill_f32<<<1, NGRAPHS, 0, stream>>>(cnt, 0.f, NGRAPHS);
  k_pool<<<(unsigned)N, HC, 0, stream>>>(accum, batch, pooled, cnt, N);
  k_mlp1<<<blocks(NGRAPHS * MLP_HID), T, 0, stream>>>(pooled, cnt, Wm1, bm1, hid);
  k_mlp2<<<1, NGRAPHS, 0, stream>>>(hid, Wm2, bm2, out);
}
